// GRU_10307921510947
// MI455X (gfx1250) — compile-verified, matrix operand order checked by reading, not run on hardware
//
#include <hip/hip_runtime.h>

// ---------------------------------------------------------------------------
// GRU LM for MI455X (gfx1250, wave32).
//  - All matmuls: v_wmma_f32_16x16x32_bf16 (f32 accumulate, bf16 operands).
//  - gi1 GEMM:    B-tile double-buffered in LDS via GLOBAL_LOAD_ASYNC_TO_LDS_B128
//                 (ASYNCcnt + split barriers).
//  - logits GEMM: B-tile double-buffered in LDS via the Tensor Data Mover
//                 (TENSOR_LOAD_TO_LDS + s_wait_tensorcnt), one issue per block.
//  - NT stores for write-once result streams (keep 192MB L2 for weights).
//  - Recurrent steps: direct-from-L2 fragment streaming (weights L2-resident).
//  - d_out's logits region doubles as scratch for the gi tensor (dead before
//    the final logits GEMM overwrites it).
// ---------------------------------------------------------------------------

typedef __attribute__((ext_vector_type(16))) __bf16        v16bf;
typedef __attribute__((ext_vector_type(8)))  float         v8f;
typedef __attribute__((ext_vector_type(4)))  unsigned int  u32x4;
typedef __attribute__((ext_vector_type(8)))  int           i32x8;
typedef __attribute__((ext_vector_type(4)))  int           i32x4;

union FragU { v16bf v; u32x4 q[2]; };

// Issue an async 16B copy global->LDS. VDST operand = LDS byte offset (low 32
// bits of the generic pointer per ISA aperture decomposition), VADDR = 64-bit
// global address. Tracked by ASYNCcnt.
#define ASYNC_LOAD_B128(ldsoff_u32, gaddr_u64)                                 \
  asm volatile("global_load_async_to_lds_b128 %0, %1, off"                     \
               :: "v"(ldsoff_u32), "v"(gaddr_u64) : "memory")

#define WAIT_ASYNCCNT(n) asm volatile("s_wait_asynccnt " #n ::: "memory")

static __device__ __forceinline__ unsigned lds_addr32(const void* p) {
  // generic LDS pointer = {shared_aperture_hi32, lds_byte_offset_lo32}
  return (unsigned)(unsigned long long)p;
}

// TDM: load a 64(row) x 32(elem) bf16 tile (row stride = K elems) into LDS at
// lds_off, rows packed contiguously (64B/row).  D# per ISA ch.8:
//   group0: count=1 | lds_addr | global_addr[56:0] | type=2
//   group1: data_size=2B, tensor_dim0=K, tensor_dim1=64, tile_dim0=32,
//           tile_dim1=64, tensor_dim0_stride=K
//   groups 2/3: zero (tile_dim2=0 -> 2D tile)
static __device__ __forceinline__ void tdm_load_64x32(unsigned long long gaddr,
                                                      unsigned lds_off, int K) {
  u32x4 g0;
  g0[0] = 1u;                                              // count=1
  g0[1] = lds_off;                                         // lds_addr
  g0[2] = (unsigned)(gaddr & 0xFFFFFFFFull);               // global_addr lo
  g0[3] = (unsigned)(gaddr >> 32) | (2u << 30);            // addr hi | type=2
  i32x8 g1;
  g1[0] = (1 << 16);                   // workgroup_mask=0 | data_size=1 (2B)
  g1[1] = (int)((unsigned)(K & 0xFFFF) << 16);             // tensor_dim0 lo16
  g1[2] = (int)(((unsigned)K >> 16) | (64u << 16));        // dim0 hi | dim1=64
  g1[3] = (int)(32u << 16);            // tensor_dim1 hi=0 | tile_dim0=32
  g1[4] = 64;                          // tile_dim1=64 | tile_dim2=0
  g1[5] = K;                           // tensor_dim0_stride lo32
  g1[6] = 0;                           // stride hi | dim1_stride lo
  g1[7] = 0;
  i32x4 z4 = {0, 0, 0, 0};
#if defined(__clang_major__) && (__clang_major__ >= 23)
  i32x8 z8 = {0, 0, 0, 0, 0, 0, 0, 0};
  __builtin_amdgcn_tensor_load_to_lds(g0, g1, z4, z4, z8, 0);
#else
  __builtin_amdgcn_tensor_load_to_lds(g0, g1, z4, z4, 0);
#endif
}

static __device__ __forceinline__ unsigned short f2bf(float x) {
  unsigned int u = __builtin_bit_cast(unsigned int, x);
  unsigned int r = u + 0x7FFFu + ((u >> 16) & 1u);   // round-to-nearest-even
  return (unsigned short)(r >> 16);
}

// 16x32 bf16 A/B fragment per-lane data: two 16B chunks at p and p+16 elems,
// where p = row_base + k + (lane>=16 ? 8 : 0).   (ISA 7.12.2 layouts)
static __device__ __forceinline__ v16bf load_frag(const unsigned short* p) {
  FragU f;
  f.q[0] = *(const u32x4*)(p);
  f.q[1] = *(const u32x4*)(p + 16);
  return f.v;
}

static __device__ __forceinline__ float sigmoidf_(float x) {
  return 1.0f / (1.0f + __expf(-x));
}

// ---------------------------------------------------------------- utilities
__global__ void k_f32_to_bf16(const float* __restrict__ src,
                              unsigned short* __restrict__ dst, int n) {
  int stride = gridDim.x * blockDim.x;
  for (int i = blockIdx.x * blockDim.x + threadIdx.x; i < n; i += stride)
    dst[i] = f2bf(src[i]);
}

__global__ void k_init_h(const float* __restrict__ src,
                         float* __restrict__ hf,
                         unsigned short* __restrict__ hbf, int n) {
  int stride = gridDim.x * blockDim.x;
  for (int i = blockIdx.x * blockDim.x + threadIdx.x; i < n; i += stride) {
    float v = src[i];
    hf[i] = v;
    hbf[i] = f2bf(v);
  }
}

__global__ void k_copy_f32(const float* __restrict__ src,
                           float* __restrict__ dst, int n) {
  int stride = gridDim.x * blockDim.x;
  for (int i = blockIdx.x * blockDim.x + threadIdx.x; i < n; i += stride)
    dst[i] = src[i];
}

// gi0[tb][g] = W_ih0[g][tok[tb]] + b_ih0[g]   (one-hot matmul == column gather)
__global__ void k_gather_gi0(const int* __restrict__ tok,
                             const float* __restrict__ Wih0,   // (3H, V)
                             const float* __restrict__ bih0,   // (3H)
                             float* __restrict__ gi, int total) {
  const int H3 = 3072, V = 8192;
  int stride = gridDim.x * blockDim.x;
  for (int i = blockIdx.x * blockDim.x + threadIdx.x; i < total; i += stride) {
    int tb = i / H3;
    int g  = i - tb * H3;
    __builtin_nontemporal_store(Wih0[(size_t)g * V + tok[tb]] + bih0[g], gi + i);
  }
}

// ------------------------------------------ WMMA GEMM, async-LDS B staging
// C[M,N] = A[M,K](bf16) @ W[N,K](bf16)^T + bias[N].
// Block: 256 threads = 8 waves; tile 128(M) x 64(N). The 64x32 B-tile for each
// K-chunk is shared by all 8 waves: double-buffered in LDS via async copies.
__global__ void k_wmma_gemm_lds(const unsigned short* __restrict__ A,
                                const unsigned short* __restrict__ W,
                                const float* __restrict__ bias,
                                float* __restrict__ C,
                                int M, int N, int K) {
  __shared__ __attribute__((aligned(16))) unsigned short sB[2][64 * 32];

  const int lane = threadIdx.x & 31;
  const int wv   = threadIdx.x >> 5;              // 0..7
  const int tiles_n = N >> 6;
  const int bm = blockIdx.x / tiles_n;            // 128-row block
  const int bn = blockIdx.x - bm * tiles_n;       // 64-col block
  const int mb = bm * 128 + wv * 16;
  const int nb = bn * 64;

  // staging: each thread async-copies one 16B slice of the 64x32 B-tile
  const int srow = threadIdx.x >> 2;              // 0..63
  const int scol = (threadIdx.x & 3) * 8;         // 0,8,16,24
  const unsigned short* wsrc = W + (size_t)(nb + srow) * K + scol;
  const unsigned sdst0 = lds_addr32(&sB[0][srow * 32 + scol]);
  const unsigned sdst1 = lds_addr32(&sB[1][srow * 32 + scol]);

  const int rsel  = lane & 15;
  const int khalf = (lane >> 4) << 3;
  const unsigned short* Ap = A + (size_t)(mb + rsel) * K + khalf;
  const unsigned short* b0p = &sB[0][(rsel +  0) * 32 + khalf];
  const unsigned short* b1p = &sB[0][(rsel + 16) * 32 + khalf];
  const unsigned short* b2p = &sB[0][(rsel + 32) * 32 + khalf];
  const unsigned short* b3p = &sB[0][(rsel + 48) * 32 + khalf];
  const int bufstride = 64 * 32;                  // elements between buffers

  v8f c0 = {}, c1 = {}, c2 = {}, c3 = {};

  ASYNC_LOAD_B128(sdst0, (unsigned long long)(const void*)wsrc);   // chunk 0
  const int nk = K >> 5;
  for (int i = 0; i < nk; ++i) {
    __syncthreads();   // all waves done reading the buffer we write next
    if (i + 1 < nk) {
      const unsigned long long g =
          (unsigned long long)(const void*)(wsrc + (size_t)(i + 1) * 32);
      if ((i & 1) == 0) { ASYNC_LOAD_B128(sdst1, g); }
      else              { ASYNC_LOAD_B128(sdst0, g); }
      WAIT_ASYNCCNT(1);    // chunk i done (in-order); chunk i+1 in flight
    } else {
      WAIT_ASYNCCNT(0);
    }
    __syncthreads();   // chunk i visible to all waves

    const int boff = (i & 1) * bufstride;
    v16bf a  = load_frag(Ap + (size_t)i * 32);
    v16bf b0 = load_frag(b0p + boff);
    v16bf b1 = load_frag(b1p + boff);
    v16bf b2 = load_frag(b2p + boff);
    v16bf b3 = load_frag(b3p + boff);
    c0 = __builtin_amdgcn_wmma_f32_16x16x32_bf16(false, a, false, b0, (short)0, c0, false, false);
    c1 = __builtin_amdgcn_wmma_f32_16x16x32_bf16(false, a, false, b1, (short)0, c1, false, false);
    c2 = __builtin_amdgcn_wmma_f32_16x16x32_bf16(false, a, false, b2, (short)0, c2, false, false);
    c3 = __builtin_amdgcn_wmma_f32_16x16x32_bf16(false, a, false, b3, (short)0, c3, false, false);
  }

  const int ncol  = lane & 15;
  const int mbase = mb + ((lane >> 4) << 3);
  const int n0 = nb + ncol;
  const float bs0 = bias ? bias[n0 +  0] : 0.0f;
  const float bs1 = bias ? bias[n0 + 16] : 0.0f;
  const float bs2 = bias ? bias[n0 + 32] : 0.0f;
  const float bs3 = bias ? bias[n0 + 48] : 0.0f;
#pragma unroll
  for (int v = 0; v < 8; ++v) {
    float* row = C + (size_t)(mbase + v) * N + n0;
    __builtin_nontemporal_store(c0[v] + bs0, row +  0);   // write-once stream:
    __builtin_nontemporal_store(c1[v] + bs1, row + 16);   // keep out of L2
    __builtin_nontemporal_store(c2[v] + bs2, row + 32);
    __builtin_nontemporal_store(c3[v] + bs3, row + 48);
  }
}

// --------------------------------------------- WMMA GEMM, TDM B staging
// Same tiling as above, but the 64x32 B-tile is DMA'd to LDS by the Tensor
// Data Mover: wave 0 issues one TENSOR_LOAD_TO_LDS per K-chunk (double-
// buffered, TENSORcnt-tracked); the other 7 waves spend issue slots only on
// the A stream and WMMAs.
__global__ void k_wmma_gemm_tdm(const unsigned short* __restrict__ A,
                                const unsigned short* __restrict__ W,
                                const float* __restrict__ bias,
                                float* __restrict__ C,
                                int M, int N, int K) {
  __shared__ __attribute__((aligned(16))) unsigned short sB[2][64 * 32];

  const int lane = threadIdx.x & 31;
  const int wv   = threadIdx.x >> 5;              // 0..7
  const int tiles_n = N >> 6;
  const int bm = blockIdx.x / tiles_n;            // 128-row block
  const int bn = blockIdx.x - bm * tiles_n;       // 64-col block
  const int mb = bm * 128 + wv * 16;
  const int nb = bn * 64;

  const unsigned long long wbase =
      (unsigned long long)(const void*)(W + (size_t)nb * K);
  const unsigned lbuf0 = lds_addr32(&sB[0][0]);
  const unsigned lbuf1 = lds_addr32(&sB[1][0]);

  const int rsel  = lane & 15;
  const int khalf = (lane >> 4) << 3;
  const unsigned short* Ap = A + (size_t)(mb + rsel) * K + khalf;
  const unsigned short* b0p = &sB[0][(rsel +  0) * 32 + khalf];
  const unsigned short* b1p = &sB[0][(rsel + 16) * 32 + khalf];
  const unsigned short* b2p = &sB[0][(rsel + 32) * 32 + khalf];
  const unsigned short* b3p = &sB[0][(rsel + 48) * 32 + khalf];
  const int bufstride = 64 * 32;

  v8f c0 = {}, c1 = {}, c2 = {}, c3 = {};

  if (wv == 0) tdm_load_64x32(wbase, lbuf0, K);   // chunk 0
  const int nk = K >> 5;
  for (int i = 0; i < nk; ++i) {
    if (wv == 0) {
      if (i + 1 < nk) {
        // buffer (i+1)&1 was last read in iter i-1; the barrier ending iter
        // i-1 ordered those reads before this issue.
        tdm_load_64x32(wbase + (unsigned long long)(i + 1) * 64,
                       ((i & 1) == 0) ? lbuf1 : lbuf0, K);
        __builtin_amdgcn_s_wait_tensorcnt(1);  // chunk i done, i+1 in flight
      } else {
        __builtin_amdgcn_s_wait_tensorcnt(0);
      }
    }
    __syncthreads();   // publish chunk i to all waves

    const int boff = (i & 1) * bufstride;
    v16bf a  = load_frag(Ap + (size_t)i * 32);
    v16bf b0 = load_frag(b0p + boff);
    v16bf b1 = load_frag(b1p + boff);
    v16bf b2 = load_frag(b2p + boff);
    v16bf b3 = load_frag(b3p + boff);
    c0 = __builtin_amdgcn_wmma_f32_16x16x32_bf16(false, a, false, b0, (short)0, c0, false, false);
    c1 = __builtin_amdgcn_wmma_f32_16x16x32_bf16(false, a, false, b1, (short)0, c1, false, false);
    c2 = __builtin_amdgcn_wmma_f32_16x16x32_bf16(false, a, false, b2, (short)0, c2, false, false);
    c3 = __builtin_amdgcn_wmma_f32_16x16x32_bf16(false, a, false, b3, (short)0, c3, false, false);

    __syncthreads();   // all reads of this buffer done before it is reused
  }

  const int ncol  = lane & 15;
  const int mbase = mb + ((lane >> 4) << 3);
  const int n0 = nb + ncol;
  const float bs0 = bias ? bias[n0 +  0] : 0.0f;
  const float bs1 = bias ? bias[n0 + 16] : 0.0f;
  const float bs2 = bias ? bias[n0 + 32] : 0.0f;
  const float bs3 = bias ? bias[n0 + 48] : 0.0f;
#pragma unroll
  for (int v = 0; v < 8; ++v) {
    float* row = C + (size_t)(mbase + v) * N + n0;
    __builtin_nontemporal_store(c0[v] + bs0, row +  0);
    __builtin_nontemporal_store(c1[v] + bs1, row + 16);
    __builtin_nontemporal_store(c2[v] + bs2, row + 32);
    __builtin_nontemporal_store(c3[v] + bs3, row + 48);
  }
}

// ------------------------------------------------------------- GRU timestep
// Wave owns one (16-batch x 16-hidden) tile; computes r/z/n tiles (3 WMMAs
// per K-chunk), then fused gate math. Ping-pong h buffers (no in-place RAW).
__global__ void k_gru_step(const unsigned short* __restrict__ hbf,  // (B,H) prev, bf16
                           const float*          __restrict__ hf,   // (B,H) prev, f32
                           const unsigned short* __restrict__ Whh,  // (3H,H) bf16
                           const float*          __restrict__ bhh,  // (3H)
                           const float*          __restrict__ gi,   // (B,3H) this step
                           float*          __restrict__ hf_out,
                           unsigned short* __restrict__ hbf_out,
                           unsigned short* __restrict__ ys_out) {   // (B,H) bf16
  const int H = 1024, H3 = 3072;
  const int lane = threadIdx.x & 31;
  const int gw = blockIdx.x * 8 + (threadIdx.x >> 5);  // 0..255
  const int tm = gw >> 6;                              // batch tile  0..3
  const int tn = gw & 63;                              // hidden tile 0..63

  const int rsel  = lane & 15;
  const int khalf = (lane >> 4) << 3;
  const unsigned short* Ap = hbf + (size_t)(tm * 16 + rsel) * H + khalf;
  const unsigned short* Wr = Whh + (size_t)(          tn * 16 + rsel) * H + khalf;
  const unsigned short* Wz = Whh + (size_t)(    H +   tn * 16 + rsel) * H + khalf;
  const unsigned short* Wn = Whh + (size_t)(2 * H +   tn * 16 + rsel) * H + khalf;

  v8f cr = {}, cz = {}, cn = {};
  for (int k = 0; k < H; k += 32) {
    v16bf a  = load_frag(Ap + k);
    v16bf br = load_frag(Wr + k);
    v16bf bz = load_frag(Wz + k);
    v16bf bn = load_frag(Wn + k);
    cr = __builtin_amdgcn_wmma_f32_16x16x32_bf16(false, a, false, br, (short)0, cr, false, false);
    cz = __builtin_amdgcn_wmma_f32_16x16x32_bf16(false, a, false, bz, (short)0, cz, false, false);
    cn = __builtin_amdgcn_wmma_f32_16x16x32_bf16(false, a, false, bn, (short)0, cn, false, false);
  }

  const int col   = tn * 16 + (lane & 15);
  const int mbase = tm * 16 + ((lane >> 4) << 3);
  const float bhr = bhh[col];
  const float bhz = bhh[H + col];
  const float bhn = bhh[2 * H + col];
#pragma unroll
  for (int v = 0; v < 8; ++v) {
    const int b = mbase + v;
    const float* gib = gi + (size_t)b * H3;
    float r = sigmoidf_(gib[col]         + cr[v] + bhr);
    float z = sigmoidf_(gib[H + col]     + cz[v] + bhz);
    float n = tanhf(    gib[2 * H + col] + r * (cn[v] + bhn));
    float hp = hf[(size_t)b * H + col];
    float hn = (1.0f - z) * n + z * hp;
    size_t o = (size_t)b * H + col;
    hf_out[o] = hn;
    unsigned short hb = f2bf(hn);
    hbf_out[o] = hb;
    ys_out[o]  = hb;
  }
}

// --------------------------------------------------------------------- host
extern "C" void kernel_launch(void* const* d_in, const int* in_sizes, int n_in,
                              void* d_out, int out_size, void* d_ws, size_t ws_size,
                              hipStream_t stream) {
  (void)in_sizes; (void)n_in; (void)out_size; (void)ws_size;
  const int V = 8192, H = 1024, T = 256, B = 64, H3 = 3 * H;

  const int*   tokens = (const int*)  d_in[0];
  const float* hidden = (const float*)d_in[1];
  const float* Wih0   = (const float*)d_in[2];
  const float* Whh0   = (const float*)d_in[3];
  const float* bih0   = (const float*)d_in[4];
  const float* bhh0   = (const float*)d_in[5];
  const float* Wih1   = (const float*)d_in[6];
  const float* Whh1   = (const float*)d_in[7];
  const float* bih1   = (const float*)d_in[8];
  const float* bhh1   = (const float*)d_in[9];
  const float* Wout   = (const float*)d_in[10];
  const float* bout   = (const float*)d_in[11];

  float* logits = (float*)d_out;                        // (T,B,V)
  float* hT_out = logits + (size_t)T * B * V;           // (L,B,H)
  float* gi     = logits;   // scratch: T*B*3H floats fit inside logits region

  // workspace layout (256B aligned), ~70 MB total
  char* ws = (char*)d_ws;
  auto alloc = [&](size_t bytes) -> char* {
    char* p = ws; ws += (bytes + 255) & ~(size_t)255; return p;
  };
  unsigned short* Whh0b = (unsigned short*)alloc((size_t)H3 * H * 2);
  unsigned short* Whh1b = (unsigned short*)alloc((size_t)H3 * H * 2);
  unsigned short* Wih1b = (unsigned short*)alloc((size_t)H3 * H * 2);
  unsigned short* Woutb = (unsigned short*)alloc((size_t)V  * H * 2);
  unsigned short* ysb   = (unsigned short*)alloc((size_t)T * B * H * 2); // ys0 then ys1
  float*          hping = (float*)         alloc((size_t)2 * B * H * 4);
  unsigned short* hbp   = (unsigned short*)alloc((size_t)2 * B * H * 2);

  const int TPB = 256;
  auto gs = [](size_t n) {
    size_t b = (n + 255) / 256; return (int)(b > 4096 ? 4096 : b);
  };

  // 1) bf16 weight conversions
  k_f32_to_bf16<<<gs((size_t)H3 * H), TPB, 0, stream>>>(Whh0, Whh0b, H3 * H);
  k_f32_to_bf16<<<gs((size_t)H3 * H), TPB, 0, stream>>>(Whh1, Whh1b, H3 * H);
  k_f32_to_bf16<<<gs((size_t)H3 * H), TPB, 0, stream>>>(Wih1, Wih1b, H3 * H);
  k_f32_to_bf16<<<gs((size_t)V  * H), TPB, 0, stream>>>(Wout, Woutb, V * H);

  // 2) gi0 = gather(W_ih0.T, tokens) + b_ih0
  k_gather_gi0<<<gs((size_t)T * B * H3), TPB, 0, stream>>>(tokens, Wih0, bih0, gi, T * B * H3);

  // 3) GRU layer 0 (256 waves/step = 32 blocks of 256)
  k_init_h<<<gs((size_t)B * H), TPB, 0, stream>>>(hidden, hping, hbp, B * H);
  for (int t = 0; t < T; ++t) {
    int s = t & 1, d = s ^ 1;
    k_gru_step<<<32, 256, 0, stream>>>(
        hbp + (size_t)s * B * H, hping + (size_t)s * B * H, Whh0b, bhh0,
        gi + (size_t)t * B * H3,
        hping + (size_t)d * B * H, hbp + (size_t)d * B * H,
        ysb + (size_t)t * B * H);
  }
  k_copy_f32<<<gs((size_t)B * H), TPB, 0, stream>>>(hping + (size_t)(T & 1) * B * H, hT_out, B * H);

  // 4) gi1 = ys0 @ W_ih1^T + b_ih1  (16384 x 3072, K=1024) — async-LDS staging
  {
    int M = T * B, N = H3;
    int blocks = (M / 128) * (N / 64);
    k_wmma_gemm_lds<<<blocks, TPB, 0, stream>>>(ysb, Wih1b, bih1, gi, M, N, H);
  }

  // 5) GRU layer 1 (ys buffer reused for ys1; gi0 already consumed)
  k_init_h<<<gs((size_t)B * H), TPB, 0, stream>>>(hidden + (size_t)B * H, hping, hbp, B * H);
  for (int t = 0; t < T; ++t) {
    int s = t & 1, d = s ^ 1;
    k_gru_step<<<32, 256, 0, stream>>>(
        hbp + (size_t)s * B * H, hping + (size_t)s * B * H, Whh1b, bhh1,
        gi + (size_t)t * B * H3,
        hping + (size_t)d * B * H, hbp + (size_t)d * B * H,
        ysb + (size_t)t * B * H);
  }
  k_copy_f32<<<gs((size_t)B * H), TPB, 0, stream>>>(hping + (size_t)(T & 1) * B * H,
                                                    hT_out + (size_t)B * H, B * H);

  // 6) logits = ys1 @ W_out^T + b_out  (16384 x 8192, K=1024) — TDM staging,
  //    overwrites gi scratch (already consumed)
  {
    int M = T * B, N = V;
    int blocks = (M / 128) * (N / 64);
    k_wmma_gemm_tdm<<<blocks, TPB, 0, stream>>>(ysb, Woutb, bout, logits, M, N, H);
  }
}
